// SacrsnSeedGPT_40948218200585
// MI455X (gfx1250) — matrix operand order repaired and checked by reading, not compiled, not run on hardware
//
#include <hip/hip_runtime.h>
#include <hip/hip_bf16.h>
#include <math.h>

// ---------------- model dims ----------------
#define VOCAB 32000
#define DM    384
#define NLAY  6
#define NEXP  4
#define NHEAD 6
#define HD    64
#define WIN   64
#define WS_S  5
#define BB    8
#define TT    256
#define SBT   (WS_S*BB*TT)     // 10240 rows through the trunk
#define BT    (BB*TT)          // 2048 rows at the head
#define DFF   (4*DM)           // 1536
#define QKVN  (3*DM)           // 1152

typedef __bf16 v16bf __attribute__((ext_vector_type(16)));
typedef __bf16 v8bf  __attribute__((ext_vector_type(8)));
typedef float  v8f   __attribute__((ext_vector_type(8)));

// =====================================================================
// fp32 -> bf16 conversion (weights, once per call)
// =====================================================================
__global__ void f2bf_kernel(const float* __restrict__ src, __bf16* __restrict__ dst, size_t n) {
    size_t i = (size_t)blockIdx.x * 256 + threadIdx.x;
    if (i < n) dst[i] = (__bf16)src[i];
}

// =====================================================================
// embedding: x[sb,t,d] = tok[idx[sb%B, t], d] + pos[t, d]
// =====================================================================
__global__ void embed_kernel(const int* __restrict__ idx, const float* __restrict__ tok,
                             const float* __restrict__ pos, float* __restrict__ x) {
    size_t i = (size_t)blockIdx.x * 256 + threadIdx.x;
    if (i >= (size_t)SBT * DM) return;
    int d  = (int)(i % DM);
    int t  = (int)((i / DM) % TT);
    int sb = (int)(i / ((size_t)DM * TT));
    int b  = sb % BB;
    int tk = idx[b * TT + t];
    x[i] = tok[(size_t)tk * DM + d] + pos[(size_t)t * DM + d];
}

// =====================================================================
// LayerNorm: one block per row of 384; writes fp32 and optional bf16
// =====================================================================
__global__ __launch_bounds__(128)
void ln_kernel(const float* __restrict__ x, const float* __restrict__ g,
               const float* __restrict__ b, float* __restrict__ outF,
               __bf16* __restrict__ outB) {
    __shared__ float s1[128], s2[128];
    const int tid = threadIdx.x;
    const size_t base = (size_t)blockIdx.x * DM;
    float s = 0.f, q = 0.f;
    for (int d = tid; d < DM; d += 128) { float v = x[base + d]; s += v; q += v * v; }
    s1[tid] = s; s2[tid] = q; __syncthreads();
    for (int o = 64; o > 0; o >>= 1) {
        if (tid < o) { s1[tid] += s1[tid + o]; s2[tid] += s2[tid + o]; }
        __syncthreads();
    }
    const float mean = s1[0] * (1.f / DM);
    const float var  = s2[0] * (1.f / DM) - mean * mean;
    const float rstd = rsqrtf(var + 1e-5f);
    for (int d = tid; d < DM; d += 128) {
        float y = (x[base + d] - mean) * rstd * g[d] + b[d];
        outF[base + d] = y;
        if (outB) outB[base + d] = (__bf16)y;
    }
}

// =====================================================================
// bf16 WMMA GEMM: C[M,N] = A[M,K] @ W[K,N] (+bias) (gelu) (*rowScale) (+=)
// 256 threads = 8 waves; tile 128(M) x 64(N), K-step 32; LDS double-buffered.
// A staged row-major [128][STA], B staged TRANSPOSED [64 n][STB k] so both
// fragments are contiguous ds_load_b128 pairs. One barrier per K-step; next
// tile's global loads issued before compute, LDS stores after. All four
// B fragments loaded before the WMMA chain -> 4 back-to-back independent
// v_wmma ops (single dscnt wait, XDL pipe stays busy).
// flags: 1 = exact GELU, 2 = accumulate into outF
// =====================================================================
#define STA 40   // padded row stride (bf16 elems): breaks LDS bank conflicts
#define STB 40

__global__ __launch_bounds__(256)
void gemm_wmma_bf16(const __bf16* __restrict__ A, const __bf16* __restrict__ W,
                    const float* __restrict__ bias, float* __restrict__ outF,
                    __bf16* __restrict__ outB, const float* __restrict__ rowScale,
                    int rsOff, int M, int N, int K, int flags) {
    __shared__ __bf16 lA[2 * 128 * STA];    // A tiles: [m][k], double buffered
    __shared__ __bf16 lB[2 * 64 * STB];     // B tiles transposed: [n][k]
    const int tid  = threadIdx.x;
    const int lane = tid & 31;
    const int wv   = tid >> 5;              // wave 0..7 -> rows wv*16..wv*16+15
    const int ln   = lane & 15;
    const int hi   = lane >> 4;
    const int m0   = blockIdx.y * 128;
    const int n0   = blockIdx.x * 64;

    // staging work assignment (fixed per thread)
    const int r0 = tid >> 2,          q0 = tid & 3;        // A chunk 0 (rows 0..63)
    const int r1 = (tid + 256) >> 2,  q1 = tid & 3;        // A chunk 1 (rows 64..127)
    const int kr0 = tid >> 5,         nc0 = tid & 31;      // B pair 0 (k rows 0..15)
    const int kr1 = kr0 + 8;                               // B pair 1 (k rows 16..31)

    v8f acc[4];
    #pragma unroll
    for (int j = 0; j < 4; ++j)
        #pragma unroll
        for (int r = 0; r < 8; ++r) acc[j][r] = 0.f;

    // ---- prologue: stage tile 0 into buffer 0 ----
    {
        const __bf16* An = A + (size_t)m0 * K;
        uint4 a0q = *(const uint4*)(An + (size_t)r0 * K + 8 * q0);
        uint4 a1q = *(const uint4*)(An + (size_t)r1 * K + 8 * q1);
        const __bf16* Wn = W + n0;
        uint32_t w00 = *(const uint32_t*)(Wn + (size_t)(2 * kr0) * N + 2 * nc0);
        uint32_t w01 = *(const uint32_t*)(Wn + (size_t)(2 * kr0 + 1) * N + 2 * nc0);
        uint32_t w10 = *(const uint32_t*)(Wn + (size_t)(2 * kr1) * N + 2 * nc0);
        uint32_t w11 = *(const uint32_t*)(Wn + (size_t)(2 * kr1 + 1) * N + 2 * nc0);
        uint32_t* dA = (uint32_t*)lA;
        *(uint4*)(dA + r0 * (STA / 2) + 4 * q0) = a0q;
        *(uint4*)(dA + r1 * (STA / 2) + 4 * q1) = a1q;
        uint32_t* dB = (uint32_t*)lB;
        dB[(2 * nc0)     * (STB / 2) + kr0] = (w00 & 0xFFFFu) | (w01 << 16);
        dB[(2 * nc0 + 1) * (STB / 2) + kr0] = (w00 >> 16)     | (w01 & 0xFFFF0000u);
        dB[(2 * nc0)     * (STB / 2) + kr1] = (w10 & 0xFFFFu) | (w11 << 16);
        dB[(2 * nc0 + 1) * (STB / 2) + kr1] = (w10 >> 16)     | (w11 & 0xFFFF0000u);
    }

    const int kIters = K >> 5;
    for (int it = 0; it < kIters; ++it) {
        __syncthreads();   // tile[it&1] visible; prior reads of tile[(it+1)&1] done

        const __bf16* cA = lA + (it & 1) * (128 * STA);
        const __bf16* cB = lB + (it & 1) * (64 * STB);
        __bf16* nA = lA + ((it + 1) & 1) * (128 * STA);
        __bf16* nB = lB + ((it + 1) & 1) * (64 * STB);

        // issue next tile's global loads up front (hidden under WMMA chain)
        const bool more = (it + 1) < kIters;
        uint4 a0q, a1q; uint32_t w00, w01, w10, w11;
        if (more) {
            const int nk0 = (it + 1) << 5;
            const __bf16* An = A + (size_t)m0 * K + nk0;
            a0q = *(const uint4*)(An + (size_t)r0 * K + 8 * q0);
            a1q = *(const uint4*)(An + (size_t)r1 * K + 8 * q1);
            const __bf16* Wn = W + (size_t)nk0 * N + n0;
            w00 = *(const uint32_t*)(Wn + (size_t)(2 * kr0) * N + 2 * nc0);
            w01 = *(const uint32_t*)(Wn + (size_t)(2 * kr0 + 1) * N + 2 * nc0);
            w10 = *(const uint32_t*)(Wn + (size_t)(2 * kr1) * N + 2 * nc0);
            w11 = *(const uint32_t*)(Wn + (size_t)(2 * kr1 + 1) * N + 2 * nc0);
        }

        // A fragment (16-bit A 16x32 layout): two contiguous 16B runs per lane
        const __bf16* ap = &cA[(wv * 16 + ln) * STA + hi * 8];
        v8bf alo = *(const v8bf*)ap;
        v8bf ahi = *(const v8bf*)(ap + 16);
        v16bf af = __builtin_shufflevector(alo, ahi, 0,1,2,3,4,5,6,7,8,9,10,11,12,13,14,15);

        // load ALL four B fragments first, then run 4 independent WMMAs
        v16bf bfr[4];
        #pragma unroll
        for (int j = 0; j < 4; ++j) {
            const __bf16* bp = &cB[(j * 16 + ln) * STB + hi * 8];
            v8bf blo = *(const v8bf*)bp;
            v8bf bhi = *(const v8bf*)(bp + 16);
            bfr[j] = __builtin_shufflevector(blo, bhi, 0,1,2,3,4,5,6,7,8,9,10,11,12,13,14,15);
        }
        #pragma unroll
        for (int j = 0; j < 4; ++j)
            acc[j] = __builtin_amdgcn_wmma_f32_16x16x32_bf16(
                false, af, false, bfr[j], (short)0, acc[j], false, false);

        // land next tile into the other LDS buffer
        if (more) {
            uint32_t* dA = (uint32_t*)nA;
            *(uint4*)(dA + r0 * (STA / 2) + 4 * q0) = a0q;
            *(uint4*)(dA + r1 * (STA / 2) + 4 * q1) = a1q;
            uint32_t* dB = (uint32_t*)nB;
            dB[(2 * nc0)     * (STB / 2) + kr0] = (w00 & 0xFFFFu) | (w01 << 16);
            dB[(2 * nc0 + 1) * (STB / 2) + kr0] = (w00 >> 16)     | (w01 & 0xFFFF0000u);
            dB[(2 * nc0)     * (STB / 2) + kr1] = (w10 & 0xFFFFu) | (w11 << 16);
            dB[(2 * nc0 + 1) * (STB / 2) + kr1] = (w10 >> 16)     | (w11 & 0xFFFF0000u);
        }
    }

    // epilogue: C/D 16x16 layout -> row = r + 8*hi, col = ln within subtile
    #pragma unroll
    for (int j = 0; j < 4; ++j) {
        #pragma unroll
        for (int r = 0; r < 8; ++r) {
            const int row = m0 + wv * 16 + r + 8 * hi;
            const int col = n0 + j * 16 + ln;
            float v = acc[j][r];
            if (bias) v += bias[col];
            if (flags & 1) v = 0.5f * v * (1.0f + erff(v * 0.70710678f));   // exact GELU
            if (rowScale) v *= rowScale[(size_t)row * NEXP + rsOff];
            const size_t o = (size_t)row * N + col;
            if (outF) { if (flags & 2) outF[o] += v; else outF[o] = v; }
            if (outB) outB[o] = (__bf16)v;
        }
    }
}

// =====================================================================
// sliding-window causal attention, one wave (32 lanes) per (sb,h,t)
// lane owns head dims {lane, lane+32}; online softmax in registers
// =====================================================================
__global__ __launch_bounds__(256)
void attn_kernel(const float* __restrict__ qkv, float* __restrict__ o) {
    const int gid  = blockIdx.x * 8 + (threadIdx.x >> 5);
    const int lane = threadIdx.x & 31;
    const int t  = gid % TT;
    const int h  = (gid / TT) % NHEAD;
    const int sb = gid / (TT * NHEAD);

    const float* qp = qkv + ((size_t)sb * TT + t) * QKVN + h * HD;
    const float q0 = qp[lane], q1 = qp[lane + 32];

    const int j0 = (t >= WIN) ? (t - WIN) : 0;
    float mx = -3.0e38f, sm = 0.f, o0 = 0.f, o1 = 0.f;
    for (int j = j0; j <= t; ++j) {
        const float* kp = qkv + ((size_t)sb * TT + j) * QKVN + DM + h * HD;
        float p = q0 * kp[lane] + q1 * kp[lane + 32];
        #pragma unroll
        for (int off = 16; off > 0; off >>= 1) p += __shfl_xor(p, off, 32);
        p *= 0.125f;                                  // 1/sqrt(64)
        const float nm = fmaxf(mx, p);
        const float c  = __expf(mx - nm);
        const float w  = __expf(p - nm);
        const float* vp = kp + DM;
        sm = sm * c + w;
        o0 = o0 * c + w * vp[lane];
        o1 = o1 * c + w * vp[lane + 32];
        mx = nm;
    }
    const float inv = 1.0f / sm;
    float* op = o + ((size_t)sb * TT + t) * DM + h * HD;
    op[lane] = o0 * inv; op[lane + 32] = o1 * inv;
}

// attention output gate + bf16 cast for proj GEMM
__global__ void gatecast_kernel(const float* __restrict__ o, const float* __restrict__ gate,
                                __bf16* __restrict__ ob) {
    size_t i = (size_t)blockIdx.x * 256 + threadIdx.x;
    if (i >= (size_t)SBT * DM) return;
    ob[i] = (__bf16)(o[i] * gate[i % DM]);
}

// =====================================================================
// MoE gate: softmax(h @ gate_w + gate_b); one wave per token row.
// Accumulates per-expert score sums for the aux balance loss.
// =====================================================================
__global__ __launch_bounds__(256)
void gate_kernel(const float* __restrict__ h, const float* __restrict__ gw,
                 const float* __restrict__ gb, float* __restrict__ sc,
                 float* __restrict__ esum) {
    const int row  = blockIdx.x * 8 + (threadIdx.x >> 5);
    const int lane = threadIdx.x & 31;
    const float* hr = h + (size_t)row * DM;
    float a0 = 0.f, a1 = 0.f, a2 = 0.f, a3 = 0.f;
    for (int d = lane; d < DM; d += 32) {
        const float v = hr[d];
        a0 += v * gw[d * 4 + 0]; a1 += v * gw[d * 4 + 1];
        a2 += v * gw[d * 4 + 2]; a3 += v * gw[d * 4 + 3];
    }
    #pragma unroll
    for (int off = 16; off > 0; off >>= 1) {
        a0 += __shfl_xor(a0, off, 32); a1 += __shfl_xor(a1, off, 32);
        a2 += __shfl_xor(a2, off, 32); a3 += __shfl_xor(a3, off, 32);
    }
    a0 += gb[0]; a1 += gb[1]; a2 += gb[2]; a3 += gb[3];
    const float mx = fmaxf(fmaxf(a0, a1), fmaxf(a2, a3));
    float e0 = __expf(a0 - mx), e1 = __expf(a1 - mx), e2 = __expf(a2 - mx), e3 = __expf(a3 - mx);
    const float inv = 1.0f / (e0 + e1 + e2 + e3);
    e0 *= inv; e1 *= inv; e2 *= inv; e3 *= inv;
    if (lane == 0) {
        float* sr = sc + (size_t)row * NEXP;
        sr[0] = e0; sr[1] = e1; sr[2] = e2; sr[3] = e3;
        atomicAdd(&esum[0], e0); atomicAdd(&esum[1], e1);
        atomicAdd(&esum[2], e2); atomicAdd(&esum[3], e3);
    }
}

// scratch: [0]=bal, [1]=ce_sum, [2]=cnt, [3..6]=expert sums
__global__ void bal_kernel(float* scratch) {
    float b = 0.f;
    for (int e = 0; e < NEXP; ++e) {
        const float m = scratch[3 + e] * (1.0f / SBT);
        b += m * m;
        scratch[3 + e] = 0.f;                         // reset for next layer
    }
    scratch[0] += b * (float)NEXP;
}

__global__ void init_scratch_kernel(float* scratch) {
    if (threadIdx.x < 8) scratch[threadIdx.x] = 0.f;
}

// =====================================================================
// final LN already done into lnx; average the S world-sims -> xf (+bf16)
// =====================================================================
__global__ void mean_kernel(const float* __restrict__ lnx, float* __restrict__ xf,
                            __bf16* __restrict__ xfb) {
    size_t i = (size_t)blockIdx.x * 256 + threadIdx.x;
    if (i >= (size_t)BT * DM) return;
    const int d = (int)(i % DM);
    const int t = (int)((i / DM) % TT);
    const int b = (int)(i / ((size_t)DM * TT));
    float s = 0.f;
    for (int si = 0; si < WS_S; ++si)
        s += lnx[(((size_t)(si * BB + b)) * TT + t) * DM + d];
    s *= (1.0f / WS_S);
    xf[i] = s;
    xfb[i] = (__bf16)s;
}

// =====================================================================
// masked cross entropy: one block per token row over V=32000 logits
// =====================================================================
__global__ __launch_bounds__(256)
void ce_kernel(const float* __restrict__ logits, const int* __restrict__ tgt,
               float* __restrict__ scratch) {
    __shared__ float red[256];
    const int tid = threadIdx.x;
    const int row = blockIdx.x;
    const float* lr = logits + (size_t)row * VOCAB;
    float m = -3.0e38f;
    for (int i = tid; i < VOCAB; i += 256) m = fmaxf(m, lr[i]);
    red[tid] = m; __syncthreads();
    for (int o = 128; o > 0; o >>= 1) {
        if (tid < o) red[tid] = fmaxf(red[tid], red[tid + o]);
        __syncthreads();
    }
    const float M = red[0]; __syncthreads();
    float s = 0.f;
    for (int i = tid; i < VOCAB; i += 256) s += __expf(lr[i] - M);
    red[tid] = s; __syncthreads();
    for (int o = 128; o > 0; o >>= 1) {
        if (tid < o) red[tid] += red[tid + o];
        __syncthreads();
    }
    if (tid == 0) {
        const int tg = tgt[row];
        if (tg != 0) {
            const float nll = logf(red[0]) + M - lr[tg];
            atomicAdd(&scratch[1], nll);
            atomicAdd(&scratch[2], 1.0f);
        }
    }
}

__global__ void finalize_kernel(const float* __restrict__ scratch, float* __restrict__ loss) {
    loss[0] = scratch[1] / fmaxf(scratch[2], 1.0f) + 0.01f * scratch[0];
}

// =====================================================================
// host-side orchestration
// =====================================================================
extern "C" void kernel_launch(void* const* d_in, const int* in_sizes, int n_in,
                              void* d_out, int out_size, void* d_ws, size_t ws_size,
                              hipStream_t stream) {
    (void)in_sizes; (void)n_in; (void)out_size; (void)ws_size;

    // ---- inputs (setup_inputs dict order) ----
    const int*   idx = (const int*)d_in[0];
    const int*   tgt = (const int*)d_in[1];
    const float* tok = (const float*)d_in[2];
    const float* pos = (const float*)d_in[3];
    auto LP = [&](int l, int k) -> const float* { return (const float*)d_in[4 + l * 15 + k]; };
    const float* lnf_g  = (const float*)d_in[4 + NLAY * 15 + 0];
    const float* lnf_b  = (const float*)d_in[4 + NLAY * 15 + 1];
    const float* head_w = (const float*)d_in[4 + NLAY * 15 + 2];
    const float* head_b = (const float*)d_in[4 + NLAY * 15 + 3];

    // ---- outputs: logits [2048,32000], loss [1], xf [2048,384] ----
    float* out      = (float*)d_out;
    float* logits   = out;
    float* loss_ptr = out + (size_t)BT * VOCAB;
    float* xf_out   = out + (size_t)BT * VOCAB + 1;

    // ---- workspace bump allocator ----
    char*  ws  = (char*)d_ws;
    size_t off = 0;
    auto alloc = [&](size_t bytes) -> char* {
        char* p = ws + off;
        off += (bytes + 255) & ~(size_t)255;
        return p;
    };
    float*  X   = (float*)alloc((size_t)SBT * DM * 4);     // residual stream
    float*  HF  = (float*)alloc((size_t)SBT * DM * 4);     // LN output fp32
    __bf16* HB  = (__bf16*)alloc((size_t)SBT * DM * 2);    // LN output bf16
    float*  QKV = (float*)alloc((size_t)SBT * QKVN * 4);
    float*  O   = (float*)alloc((size_t)SBT * DM * 4);     // attention out
    __bf16* OB  = (__bf16*)alloc((size_t)SBT * DM * 2);    // gated attn out bf16
    __bf16* HID = (__bf16*)alloc((size_t)SBT * DFF * 2);   // MoE hidden bf16
    float*  SC  = (float*)alloc((size_t)SBT * NEXP * 4);   // gate scores
    __bf16* XFB = (__bf16*)alloc((size_t)BT * DM * 2);     // xf bf16 for head
    float*  SCR = (float*)alloc(256);                      // reductions
    __bf16* WB  = (__bf16*)alloc((size_t)44138496 * 2);    // all bf16 weights

    // bf16 weight layout (element offsets)
    const size_t SZ_QKVW = (size_t)DM * QKVN;              // 442368
    const size_t SZ_PRJW = (size_t)DM * DM;                // 147456
    const size_t SZ_W1   = (size_t)NEXP * DM * DFF;        // 2359296
    const size_t SZ_W2   = (size_t)NEXP * DFF * DM;        // 2359296
    const size_t PER_L   = SZ_QKVW + SZ_PRJW + SZ_W1 + SZ_W2;
    const size_t HEAD_O  = PER_L * NLAY;
    const size_t SZ_HEAD = (size_t)DM * VOCAB;             // 12288000

    auto cvt = [&](const float* s, __bf16* d, size_t n) {
        f2bf_kernel<<<(unsigned)((n + 255) / 256), 256, 0, stream>>>(s, d, n);
    };

    init_scratch_kernel<<<1, 32, 0, stream>>>(SCR);

    // ---- convert weights to bf16 ----
    for (int l = 0; l < NLAY; ++l) {
        __bf16* lb = WB + PER_L * l;
        cvt(LP(l, 2),  lb,                              SZ_QKVW);  // qkv_w
        cvt(LP(l, 4),  lb + SZ_QKVW,                    SZ_PRJW);  // proj_w
        cvt(LP(l, 11), lb + SZ_QKVW + SZ_PRJW,          SZ_W1);    // w1 (all experts)
        cvt(LP(l, 13), lb + SZ_QKVW + SZ_PRJW + SZ_W1,  SZ_W2);    // w2 (all experts)
    }
    cvt(head_w, WB + HEAD_O, SZ_HEAD);

    // ---- embedding ----
    embed_kernel<<<(SBT * DM) / 256, 256, 0, stream>>>(idx, tok, pos, X);

    const dim3 gThr(256);
    // ---- transformer layers ----
    for (int l = 0; l < NLAY; ++l) {
        __bf16* lb = WB + PER_L * l;
        __bf16* Wqkv = lb;
        __bf16* Wprj = lb + SZ_QKVW;
        __bf16* W1b  = lb + SZ_QKVW + SZ_PRJW;
        __bf16* W2b  = lb + SZ_QKVW + SZ_PRJW + SZ_W1;

        // attention block
        ln_kernel<<<SBT, 128, 0, stream>>>(X, LP(l, 0), LP(l, 1), HF, HB);
        gemm_wmma_bf16<<<dim3(QKVN / 64, SBT / 128), gThr, 0, stream>>>(
            HB, Wqkv, LP(l, 3), QKV, nullptr, nullptr, 0, SBT, QKVN, DM, 0);
        attn_kernel<<<(SBT * NHEAD) / 8, 256, 0, stream>>>(QKV, O);
        gatecast_kernel<<<(SBT * DM) / 256, 256, 0, stream>>>(O, LP(l, 6), OB);
        gemm_wmma_bf16<<<dim3(DM / 64, SBT / 128), gThr, 0, stream>>>(
            OB, Wprj, LP(l, 5), X, nullptr, nullptr, 0, SBT, DM, DM, 2 /*acc*/);

        // MoE block
        ln_kernel<<<SBT, 128, 0, stream>>>(X, LP(l, 7), LP(l, 8), HF, HB);
        gate_kernel<<<SBT / 8, 256, 0, stream>>>(HF, LP(l, 9), LP(l, 10), SC, SCR + 3);
        bal_kernel<<<1, 1, 0, stream>>>(SCR);
        for (int e = 0; e < NEXP; ++e) {
            gemm_wmma_bf16<<<dim3(DFF / 64, SBT / 128), gThr, 0, stream>>>(
                HB, W1b + (size_t)e * DM * DFF, LP(l, 12) + (size_t)e * DFF,
                nullptr, HID, nullptr, 0, SBT, DFF, DM, 1 /*gelu*/);
            gemm_wmma_bf16<<<dim3(DM / 64, SBT / 128), gThr, 0, stream>>>(
                HID, W2b + (size_t)e * DFF * DM, LP(l, 14) + (size_t)e * DM,
                X, nullptr, SC, e, SBT, DM, DFF, 2 /*acc, scaled by score*/);
        }
    }

    // ---- final LN, world-sim mean, LM head ----
    ln_kernel<<<SBT, 128, 0, stream>>>(X, lnf_g, lnf_b, HF, nullptr);
    mean_kernel<<<(BT * DM) / 256, 256, 0, stream>>>(HF, xf_out, XFB);
    gemm_wmma_bf16<<<dim3(VOCAB / 64, BT / 128), gThr, 0, stream>>>(
        XFB, WB + HEAD_O, head_b, logits, nullptr, nullptr, 0, BT, VOCAB, DM, 0);

    // ---- loss ----
    ce_kernel<<<BT, 256, 0, stream>>>(logits, tgt, SCR);
    finalize_kernel<<<1, 1, 0, stream>>>(SCR, loss_ptr);
}
